// WordAttention_858993459764
// MI455X (gfx1250) — matrix operand order repaired and checked by reading, not compiled, not run on hardware
//
#include <hip/hip_runtime.h>
#include <hip/hip_bf16.h>

// ---------------------------------------------------------------------------
// WordAttention for MI455X (gfx1250, wave32, WMMA bf16 path)
// B=4, W=1024, c=32, C=1024, H=16, Dh=64
// ---------------------------------------------------------------------------

#define Bb   4
#define Ww   1024
#define cc   32
#define CC   1024
#define HH   16
#define DH   64
#define MM   (Bb * Ww)        // 4096 rows
#define NQKV (3 * CC)         // 3072

typedef __attribute__((ext_vector_type(16))) __bf16 bfrag;   // WMMA A/B operand
typedef __attribute__((ext_vector_type(8)))  __bf16 bhalf8;
typedef __attribute__((ext_vector_type(8)))  float  v8f;     // WMMA C/D operand

__device__ __forceinline__ v8f wmma_bf16(bfrag a, bfrag b, v8f c) {
  return __builtin_amdgcn_wmma_f32_16x16x32_bf16(
      /*neg_a=*/false, a, /*neg_b=*/false, b,
      /*c_mod=*/(short)0, c, /*reuse_a=*/false, /*reuse_b=*/false);
}

// A-matrix 16x32 bf16 fragment: lane L holds row M=L%16,
// half h=L/16 covers K = h*16 + (0..15), sequential in elements (matches B).
__device__ __forceinline__ bfrag load_a_frag(const __bf16* A, int lda,
                                             int m0, int k0, int lane) {
  int m = m0 + (lane & 15);
  int h = lane >> 4;
  const __bf16* p = A + (size_t)m * lda + k0 + h * 16;
  bhalf8 lo = *(const bhalf8*)p;          // K = k0 + h*16     .. +7
  bhalf8 hi = *(const bhalf8*)(p + 8);    // K = k0 + h*16 + 8 .. +7
  bfrag f;
#pragma unroll
  for (int i = 0; i < 8; ++i) { f[i] = lo[i]; f[i + 8] = hi[i]; }
  return f;
}

// B-matrix 32x16 bf16 fragment: lane L holds column N=n0+(L%16);
// half h=L/16 covers K = k0 + h*16 + (0..15), sequential in elements.
// Bm is [N][K] row-major, so this is one contiguous 32-byte load per lane.
__device__ __forceinline__ bfrag load_b_frag(const __bf16* Bm, int ldb,
                                             int n0, int k0, int lane) {
  int n = n0 + (lane & 15);
  int h = lane >> 4;
  return *(const bfrag*)(Bm + (size_t)n * ldb + k0 + h * 16);
}

// ---------------------------------------------------------------------------
// Stage 1: last-token index per word
// ---------------------------------------------------------------------------
__global__ void wa_last_index(const int* __restrict__ mask, int* __restrict__ lastIx) {
  int i = blockIdx.x * blockDim.x + threadIdx.x;     // word id 0..4095
  if (i < MM) {
    const int* m = mask + (size_t)i * cc;
    int s = 0;
#pragma unroll
    for (int k = 0; k < cc; ++k) s += m[k];
    lastIx[i] = s - 1;
  }
}

// ---------------------------------------------------------------------------
// Stage 2: gather last char + pos-emb; emit fp32 (output #1) and bf16 copy
// ---------------------------------------------------------------------------
__global__ void wa_pool(const float* __restrict__ x, const float* __restrict__ pos,
                        const int* __restrict__ lastIx,
                        float* __restrict__ wordF, __bf16* __restrict__ wordB) {
  size_t i = (size_t)blockIdx.x * blockDim.x + threadIdx.x;   // 0 .. 4096*1024-1
  int bw = (int)(i >> 10);
  int ch = (int)(i & (CC - 1));
  int w  = bw & (Ww - 1);
  int li = lastIx[bw];
  float v = x[((size_t)bw * cc + li) * CC + ch] + pos[(size_t)w * CC + ch];
  wordF[i] = v;
  wordB[i] = (__bf16)v;
}

// ---------------------------------------------------------------------------
// Stage 3: fp32 -> bf16 weight conversion
// ---------------------------------------------------------------------------
__global__ void wa_cvt_bf16(const float* __restrict__ in, __bf16* __restrict__ out, int n) {
  int i = blockIdx.x * blockDim.x + threadIdx.x;
  if (i < n) out[i] = (__bf16)in[i];
}

// ---------------------------------------------------------------------------
// Stage 4: QKV GEMM  qkv = word @ attn_w^T  (M=4096, N=3072, K=1024)
// 2x2 wave tile (32x32 per wave): 4 frag loads feed 4 WMMAs per K-step
// (1 load/WMMA, each fragment reused twice from registers).
// Epilogue scatters into head layouts: Q,K as [B,H,W,Dh]; V as Vt [B,H,Dh,W].
// Q is pre-scaled by 1/sqrt(Dh)=0.125.
// ---------------------------------------------------------------------------
__device__ __forceinline__ void qkv_store(const v8f& acc, int m0, int n0, int lane,
                                          __bf16* __restrict__ Qb,
                                          __bf16* __restrict__ Kb,
                                          __bf16* __restrict__ Vt) {
  int ncol = lane & 15;
  int hm   = lane >> 4;
  int e    = n0 + ncol;        // 0..3071
  int sect = e >> 10;          // 0=Q 1=K 2=V
  int ch   = e & (CC - 1);
  int h    = ch >> 6;
  int d    = ch & (DH - 1);
#pragma unroll
  for (int r = 0; r < 8; ++r) {
    int m  = m0 + hm * 8 + r;
    int b_ = m >> 10;
    int w  = m & (Ww - 1);
    float v = acc[r];
    size_t bh = (size_t)(b_ * HH + h);
    if (sect == 0)       Qb[(bh * Ww + w) * DH + d] = (__bf16)(v * 0.125f);
    else if (sect == 1)  Kb[(bh * Ww + w) * DH + d] = (__bf16)v;
    else                 Vt[(bh * DH + d) * Ww + w] = (__bf16)v;
  }
}

__global__ __launch_bounds__(128)
void wa_qkv_gemm(const __bf16* __restrict__ wordB, const __bf16* __restrict__ attnB,
                 __bf16* __restrict__ Qb, __bf16* __restrict__ Kb, __bf16* __restrict__ Vt) {
  int wave = threadIdx.x >> 5;
  int lane = threadIdx.x & 31;
  int n0 = blockIdx.x * 32;
  int m0 = (blockIdx.y * 4 + wave) * 32;

  v8f acc00 = {}, acc01 = {}, acc10 = {}, acc11 = {};
  for (int k0 = 0; k0 < CC; k0 += 32) {
    bfrag a0 = load_a_frag(wordB, CC, m0,      k0, lane);
    bfrag a1 = load_a_frag(wordB, CC, m0 + 16, k0, lane);
    bfrag b0 = load_b_frag(attnB, CC, n0,      k0, lane);
    bfrag b1 = load_b_frag(attnB, CC, n0 + 16, k0, lane);
    if (k0 + 32 < CC) {  // pull next K-slab toward the caches (global_prefetch_b8)
      __builtin_prefetch(wordB + (size_t)(m0 + (lane & 15)) * CC + k0 + 32, 0, 1);
      __builtin_prefetch(attnB + (size_t)(n0 + (lane & 15)) * CC + k0 + 32, 0, 1);
    }
    acc00 = wmma_bf16(a0, b0, acc00);
    acc01 = wmma_bf16(a0, b1, acc01);
    acc10 = wmma_bf16(a1, b0, acc10);
    acc11 = wmma_bf16(a1, b1, acc11);
  }
  qkv_store(acc00, m0,      n0,      lane, Qb, Kb, Vt);
  qkv_store(acc01, m0,      n0 + 16, lane, Qb, Kb, Vt);
  qkv_store(acc10, m0 + 16, n0,      lane, Qb, Kb, Vt);
  qkv_store(acc11, m0 + 16, n0 + 16, lane, Qb, Kb, Vt);
}

// ---------------------------------------------------------------------------
// Stage 5: causal flash attention, one wave per (b,h, 16-row q tile).
// kv processed 32 per iteration: two 16x16 S tiles (full K=32 over Dh), one
// joint online-softmax update, full 16x32 P tile staged in LDS (C->A layout
// transpose), then full-K P@V WMMAs. In-order DS + s_wait_dscnt for staging.
// ---------------------------------------------------------------------------
__global__ __launch_bounds__(32)
void wa_attention(const __bf16* __restrict__ Qb, const __bf16* __restrict__ Kb,
                  const __bf16* __restrict__ Vt, __bf16* __restrict__ Ctx) {
  int lane = threadIdx.x & 31;
  int bh   = blockIdx.x;                 // b*16 + h
  int q0   = blockIdx.y * 16;
  const __bf16* Q = Qb + (size_t)bh * Ww * DH;
  const __bf16* K = Kb + (size_t)bh * Ww * DH;
  const __bf16* V = Vt + (size_t)bh * DH * Ww;

  // Q fragments for this 16-row tile (reused across all kv tiles)
  bfrag aq0 = load_a_frag(Q, DH, q0, 0, lane);
  bfrag aq1 = load_a_frag(Q, DH, q0, 32, lane);

  v8f o0 = {}, o1 = {}, o2 = {}, o3 = {};
  float mrow[8], lrow[8];
#pragma unroll
  for (int r = 0; r < 8; ++r) { mrow[r] = -3.0e38f; lrow[r] = 0.0f; }

  __shared__ __bf16 pT[16 * 32];         // 1 KB staging tile (per-wave block)
  int ncol = lane & 15;
  int hm   = lane >> 4;

  for (int k0 = 0; k0 <= q0; k0 += 32) {
    // ---- S tiles: kv subtile 0 = k0..k0+15, subtile 1 = k0+16..k0+31
    bfrag bk00 = load_b_frag(K, DH, k0, 0, lane);
    bfrag bk01 = load_b_frag(K, DH, k0, 32, lane);
    v8f s0 = {};
    s0 = wmma_bf16(aq0, bk00, s0);
    s0 = wmma_bf16(aq1, bk01, s0);

    bool have2 = (k0 + 16 <= q0);        // uniform per wave -> EXEC stays full
    v8f s1;
    if (have2) {
      bfrag bk10 = load_b_frag(K, DH, k0 + 16, 0, lane);
      bfrag bk11 = load_b_frag(K, DH, k0 + 16, 32, lane);
      v8f t = {};
      t = wmma_bf16(aq0, bk10, t);
      t = wmma_bf16(aq1, bk11, t);
      s1 = t;
    } else {
#pragma unroll
      for (int r = 0; r < 8; ++r) s1[r] = -3.0e38f;
    }

    int kj0 = k0 + ncol;
    int kj1 = k0 + 16 + ncol;
    float tmax[8];
#pragma unroll
    for (int r = 0; r < 8; ++r) {
      int qi = q0 + hm * 8 + r;
      float v0 = s0[r]; if (kj0 > qi) v0 = -3.0e38f;
      float v1 = s1[r]; if (kj1 > qi) v1 = -3.0e38f;
      s0[r] = v0; s1[r] = v1;
      float mx = fmaxf(v0, v1);          // row-max across the 16 lanes of row
      for (int off = 1; off < 16; off <<= 1)
        mx = fmaxf(mx, __shfl_xor(mx, off, 32));
      tmax[r] = mx;
    }
#pragma unroll
    for (int r = 0; r < 8; ++r) {
      float mnew  = fmaxf(mrow[r], tmax[r]);
      float scale = __expf(mrow[r] - mnew);
      mrow[r] = mnew;
      o0[r] *= scale; o1[r] *= scale; o2[r] *= scale; o3[r] *= scale;
      float p0 = __expf(s0[r] - mnew);
      float p1 = __expf(s1[r] - mnew);
      float sum = p0 + p1;
      for (int off = 1; off < 16; off <<= 1)
        sum += __shfl_xor(sum, off, 32);
      lrow[r] = lrow[r] * scale + sum;
      s0[r] = p0; s1[r] = p1;
    }

    // ---- transpose P 16x32 (C layout) -> A layout via LDS
#pragma unroll
    for (int r = 0; r < 8; ++r) {
      pT[(hm * 8 + r) * 32 + ncol]      = (__bf16)s0[r];
      pT[(hm * 8 + r) * 32 + 16 + ncol] = (__bf16)s1[r];
    }
    asm volatile("s_wait_dscnt 0" ::: "memory");
    bfrag pa;
    {
      int m = lane & 15;
#pragma unroll
      for (int i = 0; i < 8; ++i) {
        pa[i]     = pT[m * 32 + hm * 16 + i];       // K = h*16 + 0..7
        pa[i + 8] = pT[m * 32 + hm * 16 + 8 + i];   // K = h*16 + 8 + 0..7
      }
    }
    asm volatile("s_wait_dscnt 0" ::: "memory");

    // ---- O += P(16x32) @ V(32 kv x 64); clamp tail OOB (masked lanes x P=0)
    int kc = k0 + hm * 16;
    if (kc > Ww - 16) kc = Ww - 16;
#pragma unroll
    for (int j = 0; j < 4; ++j) {
      bfrag bv = *(const bfrag*)(V + (size_t)(j * 16 + ncol) * Ww + kc);
      v8f* oj = (j == 0) ? &o0 : (j == 1) ? &o1 : (j == 2) ? &o2 : &o3;
      *oj = wmma_bf16(pa, bv, *oj);
    }
  }

  // ---- epilogue: divide by row-sum, scatter to Ctx[b, w, h*64 + d] (bf16)
  int bq = bh >> 4;
  int hh = bh & 15;
#pragma unroll
  for (int r = 0; r < 8; ++r) {
    float inv = 1.0f / lrow[r];
    int w = q0 + hm * 8 + r;
    size_t base = ((size_t)bq * Ww + w) * CC + hh * DH;
    Ctx[base +  0 + ncol] = (__bf16)(o0[r] * inv);
    Ctx[base + 16 + ncol] = (__bf16)(o1[r] * inv);
    Ctx[base + 32 + ncol] = (__bf16)(o2[r] * inv);
    Ctx[base + 48 + ncol] = (__bf16)(o3[r] * inv);
  }
}

// ---------------------------------------------------------------------------
// Stage 6: proj GEMM  out = ctx @ proj_w^T  (M=4096, N=1024, K=1024), fp32 out
// Same 2x2 wave tiling as stage 4.
// ---------------------------------------------------------------------------
__global__ __launch_bounds__(128)
void wa_proj_gemm(const __bf16* __restrict__ ctx, const __bf16* __restrict__ projB,
                  float* __restrict__ out) {
  int wave = threadIdx.x >> 5;
  int lane = threadIdx.x & 31;
  int n0 = blockIdx.x * 32;
  int m0 = (blockIdx.y * 4 + wave) * 32;

  v8f acc00 = {}, acc01 = {}, acc10 = {}, acc11 = {};
  for (int k0 = 0; k0 < CC; k0 += 32) {
    bfrag a0 = load_a_frag(ctx,   CC, m0,      k0, lane);
    bfrag a1 = load_a_frag(ctx,   CC, m0 + 16, k0, lane);
    bfrag b0 = load_b_frag(projB, CC, n0,      k0, lane);
    bfrag b1 = load_b_frag(projB, CC, n0 + 16, k0, lane);
    acc00 = wmma_bf16(a0, b0, acc00);
    acc01 = wmma_bf16(a0, b1, acc01);
    acc10 = wmma_bf16(a1, b0, acc10);
    acc11 = wmma_bf16(a1, b1, acc11);
  }
  int ncol = lane & 15;
  int hm   = lane >> 4;
#pragma unroll
  for (int r = 0; r < 8; ++r) {
    int ma = m0 + hm * 8 + r;
    int mb = m0 + 16 + hm * 8 + r;
    out[(size_t)ma * CC + n0 + ncol]      = acc00[r];
    out[(size_t)ma * CC + n0 + 16 + ncol] = acc01[r];
    out[(size_t)mb * CC + n0 + ncol]      = acc10[r];
    out[(size_t)mb * CC + n0 + 16 + ncol] = acc11[r];
  }
}

// ---------------------------------------------------------------------------
// Host-side launcher
// ---------------------------------------------------------------------------
extern "C" void kernel_launch(void* const* d_in, const int* in_sizes, int n_in,
                              void* d_out, int out_size, void* d_ws, size_t ws_size,
                              hipStream_t stream) {
  const float* x      = (const float*)d_in[0];   // [B, W, c, C]
  const float* pos    = (const float*)d_in[1];   // [W, C]
  const float* attn_w = (const float*)d_in[2];   // [3C, C]
  const float* proj_w = (const float*)d_in[3];   // [C, C]
  const int*   mask   = (const int*)  d_in[4];   // [B, W, c]

  float* wordF = (float*)d_out;                          // output #1: word [B,W,C]
  float* outF  = (float*)d_out + (size_t)MM * CC;        // output #2: out  [B,W,C]

  // workspace carve-up (bf16 working set ~48 MB)
  char* ws = (char*)d_ws;
  int*    lastIx = (int*)ws;                                       // 16 KB
  __bf16* wordB  = (__bf16*)(ws + 16384);                          // 4096*1024
  __bf16* attnB  = wordB + (size_t)MM * CC;                        // 3072*1024
  __bf16* projB  = attnB + (size_t)NQKV * CC;                      // 1024*1024
  __bf16* Qb     = projB + (size_t)CC * CC;                        // [B,H,W,Dh]
  __bf16* Kb     = Qb    + (size_t)MM * CC;
  __bf16* Vt     = Kb    + (size_t)MM * CC;                        // [B,H,Dh,W]
  __bf16* Ctx    = Vt    + (size_t)MM * CC;                        // [B,W,C]

  // 1) last-token index
  wa_last_index<<<dim3((MM + 255) / 256), dim3(256), 0, stream>>>(mask, lastIx);
  // 2) gather + pos emb (fp32 output + bf16 copy)
  wa_pool<<<dim3((MM * CC) / 256), dim3(256), 0, stream>>>(x, pos, lastIx, wordF, wordB);
  // 3) weights -> bf16
  wa_cvt_bf16<<<dim3((NQKV * CC) / 256), dim3(256), 0, stream>>>(attn_w, attnB, NQKV * CC);
  wa_cvt_bf16<<<dim3((CC * CC) / 256), dim3(256), 0, stream>>>(proj_w, projB, CC * CC);
  // 4) QKV GEMM: grid (N/32, M/32/4 waves)
  wa_qkv_gemm<<<dim3(NQKV / 32, MM / 32 / 4), dim3(128), 0, stream>>>(wordB, attnB, Qb, Kb, Vt);
  // 5) flash attention: grid (B*H, W/16), one wave per block
  wa_attention<<<dim3(Bb * HH, Ww / 16), dim3(32), 0, stream>>>(Qb, Kb, Vt, Ctx);
  // 6) projection GEMM
  wa_proj_gemm<<<dim3(CC / 32, MM / 32 / 4), dim3(128), 0, stream>>>(Ctx, projB, outF);
}
